// SGConv_39857296507459
// MI455X (gfx1250) — matrix operand order repaired and checked by reading, compiled, not run on hardware
//
#include <hip/hip_runtime.h>
#include <hip/hip_bf16.h>
#include <cmath>

typedef __attribute__((ext_vector_type(16))) _Float16 v16h;
typedef __attribute__((ext_vector_type(8)))  _Float16 v8h;
typedef __attribute__((ext_vector_type(8)))  float    v8f;
typedef __attribute__((ext_vector_type(4)))  int      v4i;

constexpr int   Bb  = 2;
constexpr int   Nn  = 8192;
constexpr int   Hh  = 128;
constexpr float EPS = 1e-6f;

// ---------------------------------------------------------------------------
// Async global->LDS copy (CDNA5 GLOBAL_LOAD_ASYNC_TO_LDS_B128, ASYNCcnt).
// ---------------------------------------------------------------------------
#if __has_builtin(__builtin_amdgcn_global_load_async_to_lds_b128)
#define HAVE_ASYNC_LDS 1
typedef __attribute__((address_space(1))) v4i g_v4i;   // global memory
typedef __attribute__((address_space(3))) v4i l_v4i;   // LDS
#else
#define HAVE_ASYNC_LDS 0
#endif

__device__ inline void async_copy16(const void* gsrc, void* ldst) {
#if HAVE_ASYNC_LDS
    __builtin_amdgcn_global_load_async_to_lds_b128((g_v4i*)gsrc, (l_v4i*)ldst, 0, 0);
#else
    *(uint4*)ldst = *(const uint4*)gsrc;
#endif
}

__device__ inline void wait_async_copies() {
#if HAVE_ASYNC_LDS
#if __has_builtin(__builtin_amdgcn_s_wait_asynccnt)
    __builtin_amdgcn_s_wait_asynccnt(0);
#else
    asm volatile("s_wait_asynccnt 0x0" ::: "memory");
#endif
#endif
    // fallback path: ds stores are ordered by the barrier's dscnt wait
}

// ---------------------------------------------------------------------------
// Kernel A: in_degrees = sum|adj| per row -> norm = (deg+eps)^-1/2   [B*N]
// ---------------------------------------------------------------------------
__global__ void __launch_bounds__(256)
degnorm_kernel(const float* __restrict__ adj, float* __restrict__ norm) {
    const int row = blockIdx.x;            // 0 .. B*N-1
    const int tid = threadIdx.x;
    const float4* p = (const float4*)(adj + (size_t)row * Nn);
    float s = 0.f;
#pragma unroll
    for (int i = 0; i < 8; ++i) {
        float4 v = p[i * 256 + tid];
        s += fabsf(v.x) + fabsf(v.y) + fabsf(v.z) + fabsf(v.w);
    }
    __shared__ float red[256];
    red[tid] = s;
    __syncthreads();
    for (int off = 128; off > 0; off >>= 1) {
        if (tid < off) red[tid] += red[tid + off];
        __syncthreads();
    }
    if (tid == 0) norm[row] = 1.0f / sqrtf(red[0] + EPS);
}

// ---------------------------------------------------------------------------
// Kernel B: snT[b][h][n] = f16( norm[b,n] * (x[b,n,:] . W[:,h]) )
// ---------------------------------------------------------------------------
__global__ void __launch_bounds__(256)
support_kernel(const float* __restrict__ x, const float* __restrict__ W,
               const float* __restrict__ norm, _Float16* __restrict__ snT) {
    const int b   = blockIdx.y;
    const int nb  = blockIdx.x * 16;
    const int tid = threadIdx.x;

    __shared__ float xs[16][130];
#pragma unroll
    for (int i = 0; i < 2; ++i) {
        int f  = i * 256 + tid;
        int r  = f >> 5;
        int c4 = f & 31;
        float4 v = *(const float4*)(x + (size_t)(b * Nn + nb + r) * Hh + c4 * 4);
        xs[r][c4 * 4 + 0] = v.x; xs[r][c4 * 4 + 1] = v.y;
        xs[r][c4 * 4 + 2] = v.z; xs[r][c4 * 4 + 3] = v.w;
    }
    __syncthreads();

    const int   j  = tid & 15;
    const int   h0 = tid >> 4;
    const float nm = norm[b * Nn + nb + j];
#pragma unroll
    for (int hh = 0; hh < 8; ++hh) {
        const int h = hh * 16 + h0;
        float s = 0.f;
        for (int k = 0; k < Hh; ++k) s = fmaf(xs[j][k], W[k * Hh + h], s);
        snT[(size_t)(b * Hh + h) * Nn + nb + j] = (_Float16)(s * nm);
    }
}

// ---------------------------------------------------------------------------
// Kernel C: out[b,m,h] = elu( norm[b,m] * sum_n adj[b,m,n]*snT[b,h,n] + bias[h] )
// 128x128 tile / WG, 128 threads (4 waves), 2 M-strips per wave so each B
// fragment feeds two WMMAs (1.5 ds_load_b128 per WMMA -> under the LDS roof).
// f16 WMMA 16x16x32 w/ f32 accum; K tile 64; double-buffered async copies.
// ---------------------------------------------------------------------------
constexpr int KB    = 64;   // K tile
constexpr int LDA_A = 68;   // adj LDS row stride (f32 words): (m*17)%16 walk, conflict-free
constexpr int LDB   = 72;   // snT LDS row stride (halves):    (n*9)%16 walk, conflict-free

__device__ inline v16h cat16(v8h lo, v8h hi) {
    v16h r;
#pragma unroll
    for (int i = 0; i < 8; ++i) { r[i] = lo[i]; r[i + 8] = hi[i]; }
    return r;
}

__global__ void __launch_bounds__(128, 1)
spmm_wmma_kernel(const float* __restrict__ adj, const _Float16* __restrict__ snT,
                 const float* __restrict__ norm, const float* __restrict__ bias,
                 float* __restrict__ out) {
    const int b     = blockIdx.y;
    const int mBase = blockIdx.x * 128;
    const int tid   = threadIdx.x;         // 0..127, 4 waves
    const int wv    = tid >> 5;            // wave id -> strips {wv, wv+4}
    const int lane  = tid & 31;
    const int hi    = lane >> 4;
    const int lo16  = lane & 15;

    __shared__ float    Abuf[2][128 * LDA_A];  // adj tile, f32  128 x 64
    __shared__ _Float16 Bbuf[2][128 * LDB];    // snT tile, f16  128(h) x 64(k)

    const float* adjB = adj + (size_t)b * Nn * Nn;

    v8f acc[2][8];
    {
        v8f z;
#pragma unroll
        for (int i = 0; i < 8; ++i) z[i] = 0.f;
#pragma unroll
        for (int p = 0; p < 2; ++p)
#pragma unroll
            for (int t = 0; t < 8; ++t) acc[p][t] = z;
    }

    // Issue one tile's worth of async global->LDS copies (no staging VGPRs).
    auto copyTile = [&](int kt, int buf) {
        const int kBase = kt * KB;
#pragma unroll
        for (int i = 0; i < 16; ++i) {         // adj: 2048 x 16B
            int f = i * 128 + tid;
            int r = f >> 4, c4 = f & 15;
            async_copy16(adjB + (size_t)(mBase + r) * Nn + kBase + c4 * 4,
                         &Abuf[buf][r * LDA_A + c4 * 4]);
        }
#pragma unroll
        for (int i = 0; i < 8; ++i) {          // snT: 1024 x 16B
            int f = i * 128 + tid;
            int r = f >> 3, sg = f & 7;
            async_copy16(snT + (size_t)(b * Hh + r) * Nn + kBase + sg * 8,
                         &Bbuf[buf][r * LDB + sg * 8]);
        }
    };
    // A fragment (f32 in LDS -> f16): lane(l): M = l%16, K = g*16+8*hi+idx
    auto loadA = [&](int cb, int kk, int strip) -> v16h {
        const float* Ap = &Abuf[cb][(strip * 16 + lo16) * LDA_A + kk * 32 + hi * 8];
        float4 a0 = *(const float4*)(Ap);
        float4 a1 = *(const float4*)(Ap + 4);
        float4 a2 = *(const float4*)(Ap + 16);
        float4 a3 = *(const float4*)(Ap + 20);
        v16h af;
        af[0]  = (_Float16)a0.x; af[1]  = (_Float16)a0.y;
        af[2]  = (_Float16)a0.z; af[3]  = (_Float16)a0.w;
        af[4]  = (_Float16)a1.x; af[5]  = (_Float16)a1.y;
        af[6]  = (_Float16)a1.z; af[7]  = (_Float16)a1.w;
        af[8]  = (_Float16)a2.x; af[9]  = (_Float16)a2.y;
        af[10] = (_Float16)a2.z; af[11] = (_Float16)a2.w;
        af[12] = (_Float16)a3.x; af[13] = (_Float16)a3.y;
        af[14] = (_Float16)a3.z; af[15] = (_Float16)a3.w;
        return af;
    };
    // B fragment: lane(l): N = l%16, K = g*16+8*hi+idx
    auto loadB = [&](int cb, int kk, int t) -> v16h {
        const _Float16* Bp = &Bbuf[cb][(t * 16 + lo16) * LDB + kk * 32 + hi * 8];
        return cat16(*(const v8h*)Bp, *(const v8h*)(Bp + 16));
    };

    copyTile(0, 0);
    wait_async_copies();
    __syncthreads();

    int cur = 0;
    for (int kt = 0; kt < Nn / KB; ++kt) {
        const bool more = (kt + 1) < (Nn / KB);
        if (more) copyTile(kt + 1, cur ^ 1);   // overlaps with compute below

#pragma unroll
        for (int kk = 0; kk < 2; ++kk) {
            v16h af0 = loadA(cur, kk, wv);
            v16h af1 = loadA(cur, kk, wv + 4);
            v16h bfr[8];
#pragma unroll
            for (int t = 0; t < 8; ++t) bfr[t] = loadB(cur, kk, t);
#pragma unroll
            for (int t = 0; t < 8; ++t) {
                acc[0][t] = __builtin_amdgcn_wmma_f32_16x16x32_f16(
                    false, af0, false, bfr[t], (short)0, acc[0][t], false, false);
                acc[1][t] = __builtin_amdgcn_wmma_f32_16x16x32_f16(
                    false, af1, false, bfr[t], (short)0, acc[1][t], false, false);
            }
        }

        if (more) wait_async_copies();         // own copies landed in LDS
        __syncthreads();                       // visible to all waves
        cur ^= 1;
    }

    // Epilogue: C/D layout -> M = v + 8*hi, N(=h) = l%16
#pragma unroll
    for (int p = 0; p < 2; ++p) {
        const int sBase = mBase + (wv + 4 * p) * 16;
        float nm[8];
#pragma unroll
        for (int v = 0; v < 8; ++v)
            nm[v] = norm[b * Nn + sBase + v + 8 * hi];
#pragma unroll
        for (int t = 0; t < 8; ++t) {
            const int   h  = t * 16 + lo16;
            const float bs = bias[h];
#pragma unroll
            for (int v = 0; v < 8; ++v) {
                const int m  = sBase + v + 8 * hi;
                float val = acc[p][t][v] * nm[v] + bs;
                val = val > 0.f ? val : (expf(val) - 1.f);   // ELU, alpha=1
                out[(size_t)(b * Nn + m) * Hh + h] = val;
            }
        }
    }
}

// ---------------------------------------------------------------------------
extern "C" void kernel_launch(void* const* d_in, const int* in_sizes, int n_in,
                              void* d_out, int out_size, void* d_ws, size_t ws_size,
                              hipStream_t stream) {
    (void)in_sizes; (void)n_in; (void)out_size; (void)ws_size;
    const float* x    = (const float*)d_in[0];
    const float* adj  = (const float*)d_in[1];
    const float* W    = (const float*)d_in[2];
    const float* bias = (const float*)d_in[3];
    float*       out  = (float*)d_out;

    // workspace: norm [B*N] f32 (64 KB), then snT [B*H*N] f16 (4 MB)
    float*    norm = (float*)d_ws;
    _Float16* snT  = (_Float16*)((char*)d_ws + 64 * 1024);

    degnorm_kernel  <<<Bb * Nn,             256, 0, stream>>>(adj, norm);
    support_kernel  <<<dim3(Nn / 16, Bb),   256, 0, stream>>>(x, W, norm, snT);
    spmm_wmma_kernel<<<dim3(Nn / 128, Bb),  128, 0, stream>>>(adj, snT, norm, bias, out);
}
